// Attention_16054587752919
// MI455X (gfx1250) — matrix operand order repaired and checked by reading, not compiled, not run on hardware
//
#include <hip/hip_runtime.h>

// ---------------- problem constants ----------------
#define BN    2
#define TN    2048
#define DN    2048
#define NQ    16
#define KVH   8
#define HD    128
#define WINSZ 1024
#define SOFTCAP 50.0f
#define QSCALAR 0.08838834764831845f

typedef _Float16 h16;
typedef __attribute__((ext_vector_type(16))) _Float16 v16h;
typedef __attribute__((ext_vector_type(8)))  float    v8f;

union Frag { unsigned u[8]; v16h h; };

__device__ __forceinline__ unsigned pack2(float a, float b) {
  union { h16 h[2]; unsigned u; } v;
  v.h[0] = (h16)a; v.h[1] = (h16)b;
  return v.u;
}

__device__ __forceinline__ v8f wmma_f16(const Frag& a, const Frag& b, v8f c) {
  return __builtin_amdgcn_wmma_f32_16x16x32_f16(false, a.h, false, b.h,
                                                (short)0, c, false, false);
}

// =====================================================================
// Kernel 1: fused QKV projection (+RoPE, +query scale), f32 -> f16 out
// grid: (Ttile=16, head=32, b=2), block 256 (8 waves)
// head 0..15 -> q (wq), 16..23 -> k (wkv[0]), 24..31 -> v (wkv[1])
// =====================================================================
__global__ __launch_bounds__(256) void qkv_kernel(
    const float* __restrict__ x, const float* __restrict__ wq,
    const float* __restrict__ wkv, const int* __restrict__ segpos,
    h16* __restrict__ qh, h16* __restrict__ kh, h16* __restrict__ vh)
{
  const int tTile = blockIdx.x, head = blockIdx.y, b = blockIdx.z;
  const int tid = threadIdx.x, wave = tid >> 5, lane = tid & 31;
  const int l16 = lane & 15, half = lane >> 4;
  const int t0 = tTile * 128;

  __shared__ unsigned Alds[128 * 17];   // [row][k/2], stride 17 dwords (34 h16)
  __shared__ unsigned Btlds[128 * 17];  // [col(h)][k/2]

  const float* wsrc;
  if (head < 16)      wsrc = wq  + (size_t)head * DN * HD;
  else if (head < 24) wsrc = wkv + (size_t)(head - 16) * DN * HD;
  else                wsrc = wkv + (size_t)(KVH + head - 24) * DN * HD;

  v8f acc[8];
  const v8f vz = {0.f,0.f,0.f,0.f,0.f,0.f,0.f,0.f};
  for (int j = 0; j < 8; ++j) acc[j] = vz;

  for (int kk = 0; kk < DN; kk += 32) {
    __syncthreads();
    // stage A: x[b, t0+row, kk+k]  (128 x 32 f32) -> f16 LDS
    for (int it = 0; it < 4; ++it) {
      int idx = tid + it * 256;            // 0..1023 float4's
      int row = idx >> 3;
      int k4  = (idx & 7) * 4;
      const float* px = x + ((size_t)(b * TN + t0 + row)) * DN + kk + k4;
      float4 f = *(const float4*)px;
      if (kk + 32 < DN) __builtin_prefetch(px + 32, 0, 0);
      Alds[row * 17 + (k4 >> 1)    ] = pack2(f.x, f.y);
      Alds[row * 17 + (k4 >> 1) + 1] = pack2(f.z, f.w);
    }
    // stage B transposed: w[kk+i][h] -> Bt[h][i]
    for (int it = 0; it < 4; ++it) {
      int idx = tid + it * 256;
      int i  = idx >> 5;
      int h4 = (idx & 31) * 4;
      const float* pw = wsrc + (size_t)(kk + i) * HD + h4;
      float4 f = *(const float4*)pw;
      h16* bt = (h16*)Btlds;
      bt[(h4 + 0) * 34 + i] = (h16)f.x;
      bt[(h4 + 1) * 34 + i] = (h16)f.y;
      bt[(h4 + 2) * 34 + i] = (h16)f.z;
      bt[(h4 + 3) * 34 + i] = (h16)f.w;
    }
    __syncthreads();

    // A fragment: lane = row, K pairs per ISA 16-bit A layout
    Frag a;
    const int arow = wave * 16 + l16;
    for (int j = 0; j < 4; ++j) a.u[j]     = Alds[arow * 17 + half * 4 + j];
    for (int j = 0; j < 4; ++j) a.u[4 + j] = Alds[arow * 17 + 8 + half * 4 + j];

    for (int ct = 0; ct < 8; ++ct) {
      Frag bf;
      const int col = ct * 16 + l16;
      for (int j = 0; j < 8; ++j) bf.u[j] = Btlds[col * 17 + (j >> 2) * 8 + half * 4 + (j & 3)];
      acc[ct] = wmma_f16(a, bf, acc[ct]);
    }
  }

  // epilogue
  const int rowbase = t0 + wave * 16 + half * 8;
  if (head < 24) {  // q or k: RoPE
    const bool isq = head < 16;
    h16* dst = isq ? (qh + ((size_t)(b * NQ  + head)        * TN) * HD)
                   : (kh + ((size_t)(b * KVH + (head - 16)) * TN) * HD);
    const float scal = isq ? QSCALAR : 1.0f;
    for (int r = 0; r < 8; ++r) {
      const int trow = rowbase + r;
      const float pos = (float)segpos[b * TN + trow];
      for (int j = 0; j < 4; ++j) {
        const int h1 = j * 16 + l16;
        float invf = __powf(10000.0f, -(float)h1 * (1.0f / 64.0f));
        float ang = pos * invf;
        float s, c;
        __sincosf(ang, &s, &c);
        float f1 = acc[j][r], f2 = acc[j + 4][r];
        float o1 = (f1 * c - f2 * s) * scal;
        float o2 = (f2 * c + f1 * s) * scal;
        dst[(size_t)trow * HD + h1     ] = (h16)o1;
        dst[(size_t)trow * HD + h1 + 64] = (h16)o2;
      }
    }
  } else {          // v: plain
    h16* dst = vh + ((size_t)(b * KVH + (head - 24)) * TN) * HD;
    for (int r = 0; r < 8; ++r) {
      const int trow = rowbase + r;
      for (int j = 0; j < 8; ++j)
        dst[(size_t)trow * HD + j * 16 + l16] = (h16)acc[j][r];
    }
  }
}

// =====================================================================
// Kernel 2: flash attention, sliding-window causal, soft cap, GQA (G=2)
// grid: (Ttile=16, n=16, b=2), block 256 (8 waves, 16 query rows each)
// =====================================================================
__global__ __launch_bounds__(256) void attn_kernel(
    const h16* __restrict__ qh, const h16* __restrict__ kh,
    const h16* __restrict__ vh, const unsigned char* __restrict__ amask,
    h16* __restrict__ enc)
{
  const int tTile = blockIdx.x, n = blockIdx.y, b = blockIdx.z;
  const int kvh = n >> 1;                      // G = NQ/KVH = 2
  const int tid = threadIdx.x, wave = tid >> 5, lane = tid & 31;
  const int l16 = lane & 15, half = lane >> 4;
  const int t0 = tTile * 128;

  __shared__ unsigned Klds[32 * 68];    // [key][h/2], stride 68 dwords
  __shared__ unsigned Vtlds[128 * 68];  // [h][key/2], stride 68 dwords
  __shared__ unsigned Plds[8][16 * 17]; // per-wave P scratch

  // preload Q fragments (A layout), 4 K-chunks of 32 over H=128
  const unsigned* qp =
      (const unsigned*)(qh + ((size_t)(b * NQ + n) * TN) * HD);
  const int qrow = t0 + wave * 16 + l16;
  Frag qf[4];
  for (int c = 0; c < 4; ++c) {
    for (int j = 0; j < 4; ++j)
      qf[c].u[j]     = qp[(size_t)qrow * 64 + c * 16 + half * 4 + j];
    for (int j = 0; j < 4; ++j)
      qf[c].u[4 + j] = qp[(size_t)qrow * 64 + c * 16 + 8 + half * 4 + j];
  }

  v8f acc[8];
  const v8f vz = {0.f,0.f,0.f,0.f,0.f,0.f,0.f,0.f};
  for (int j = 0; j < 8; ++j) acc[j] = vz;
  float mrow[8], srow[8];
  for (int r = 0; r < 8; ++r) { mrow[r] = -1e30f; srow[r] = 0.f; }

  const int sEnd = t0 + 128;
  int sBeg = t0 + 127 - (WINSZ - 1);
  if (sBeg < 0) sBeg = 0;
  sBeg &= ~31;

  const h16* kbase = kh + ((size_t)(b * KVH + kvh) * TN) * HD;
  const h16* vbase = vh + ((size_t)(b * KVH + kvh) * TN) * HD;
  const int rbase = t0 + wave * 16 + half * 8;

  for (int s0 = sBeg; s0 < sEnd; s0 += 32) {
    __syncthreads();
    // stage K block (32 keys x 128 h f16)
    const unsigned* kg = (const unsigned*)(kbase + (size_t)s0 * HD);
    for (int it = 0; it < 8; ++it) {
      int idx = tid + it * 256;
      int key = idx >> 6, hd = idx & 63;
      Klds[key * 68 + hd] = kg[key * 64 + hd];
    }
    // stage V transposed (h-major so PV B-fragments are contiguous pairs)
    const unsigned* vg = (const unsigned*)(vbase + (size_t)s0 * HD);
    h16* vt = (h16*)Vtlds;
    for (int it = 0; it < 8; ++it) {
      int idx = tid + it * 256;
      int key = idx >> 6, hd = idx & 63;
      unsigned d = vg[key * 64 + hd];
      union { unsigned short s; h16 h; } lo, hi;
      lo.s = (unsigned short)(d & 0xffffu);
      hi.s = (unsigned short)(d >> 16);
      vt[(hd * 2 + 0) * 136 + key] = lo.h;
      vt[(hd * 2 + 1) * 136 + key] = hi.h;
    }
    __syncthreads();

    // S = Q @ K^T  (two 16x16 tiles over 32 keys, K-dim = H = 4 wmma each)
    v8f sv[2];
    for (int tI = 0; tI < 2; ++tI) {
      v8f s = vz;
      const int keyc = tI * 16 + l16;
      for (int c = 0; c < 4; ++c) {
        Frag bf;
        for (int j = 0; j < 8; ++j)
          bf.u[j] = Klds[keyc * 68 + c * 16 + (j >> 2) * 8 + half * 4 + (j & 3)];
        s = wmma_f16(qf[c], bf, s);
      }
      sv[tI] = s;
    }

    // soft cap + mask + block row-max
    float bmax[8];
    for (int r = 0; r < 8; ++r) bmax[r] = -1e30f;
    for (int tI = 0; tI < 2; ++tI) {
      const int sk = s0 + tI * 16 + l16;
      for (int r = 0; r < 8; ++r) {
        const int tq = rbase + r;
        float lg = sv[tI][r];
        float z = lg * (1.0f / SOFTCAP);
        z = fminf(fmaxf(z, -15.f), 15.f);
        float e2 = __expf(2.f * z);
        lg = SOFTCAP * ((e2 - 1.f) / (e2 + 1.f));
        bool valid = (sk > tq - WINSZ) &&
                     (amask[(size_t)(b * TN + tq) * TN + sk] != 0);
        lg = valid ? lg : -1e30f;
        sv[tI][r] = lg;
        bmax[r] = fmaxf(bmax[r], lg);
      }
    }
    for (int r = 0; r < 8; ++r) {
      float v = bmax[r];
      v = fmaxf(v, __shfl_xor(v, 1, 32));
      v = fmaxf(v, __shfl_xor(v, 2, 32));
      v = fmaxf(v, __shfl_xor(v, 4, 32));
      v = fmaxf(v, __shfl_xor(v, 8, 32));
      bmax[r] = v;
    }
    float alpha[8];
    for (int r = 0; r < 8; ++r) {
      float mnew = fmaxf(mrow[r], bmax[r]);
      alpha[r] = __expf(mrow[r] - mnew);
      mrow[r] = mnew;
    }
    float psum[8];
    for (int r = 0; r < 8; ++r) psum[r] = 0.f;
    for (int tI = 0; tI < 2; ++tI)
      for (int r = 0; r < 8; ++r) {
        float lg = sv[tI][r];
        float p = (lg > -0.5e30f) ? __expf(lg - mrow[r]) : 0.f;
        sv[tI][r] = p;
        psum[r] += p;
      }
    for (int r = 0; r < 8; ++r) {
      float v = psum[r];
      v += __shfl_xor(v, 1, 32);
      v += __shfl_xor(v, 2, 32);
      v += __shfl_xor(v, 4, 32);
      v += __shfl_xor(v, 8, 32);
      srow[r] = srow[r] * alpha[r] + v;
    }
    for (int j = 0; j < 8; ++j)
      for (int r = 0; r < 8; ++r) acc[j][r] *= alpha[r];

    // P: C-layout -> A-layout via per-wave LDS, then PV
    unsigned* pl = Plds[wave];
    h16* ph = (h16*)pl;
    for (int tI = 0; tI < 2; ++tI)
      for (int r = 0; r < 8; ++r)
        ph[(half * 8 + r) * 34 + tI * 16 + l16] = (h16)sv[tI][r];
    Frag pa;
    for (int j = 0; j < 4; ++j) pa.u[j]     = pl[l16 * 17 + half * 4 + j];
    for (int j = 0; j < 4; ++j) pa.u[4 + j] = pl[l16 * 17 + 8 + half * 4 + j];
    for (int ht = 0; ht < 8; ++ht) {
      Frag vf;
      const int hc = ht * 16 + l16;
      for (int j = 0; j < 8; ++j) vf.u[j] = Vtlds[hc * 68 + (j >> 2) * 8 + half * 4 + (j & 3)];
      acc[ht] = wmma_f16(pa, vf, acc[ht]);
    }
  }

  // normalize and write encoded [B,T,N*H] as f16
  h16* dst = enc + (size_t)b * TN * (NQ * HD);
  for (int r = 0; r < 8; ++r) {
    float inv = (srow[r] > 0.f) ? (1.0f / srow[r]) : 0.f;
    const int trow = rbase + r;
    for (int j = 0; j < 8; ++j)
      dst[(size_t)trow * (NQ * HD) + n * HD + j * 16 + l16] =
          (h16)(acc[j][r] * inv);
  }
}

// =====================================================================
// Kernel 3: output projection  out[4096 x 2048] = enc(f16) @ wo(f32)
// grid: (rowTile=32, colTile=16), block 256 (8 waves)
// =====================================================================
__global__ __launch_bounds__(256) void outproj_kernel(
    const h16* __restrict__ enc, const float* __restrict__ wo,
    float* __restrict__ out)
{
  const int rt = blockIdx.x, ct = blockIdx.y;
  const int tid = threadIdx.x, wave = tid >> 5, lane = tid & 31;
  const int l16 = lane & 15, half = lane >> 4;
  const int r0 = rt * 128, c0 = ct * 128;

  __shared__ unsigned Alds[128 * 17];
  __shared__ unsigned Btlds[128 * 17];

  v8f acc[8];
  const v8f vz = {0.f,0.f,0.f,0.f,0.f,0.f,0.f,0.f};
  for (int j = 0; j < 8; ++j) acc[j] = vz;

  const unsigned* ep = (const unsigned*)enc;  // rows of 1024 dwords
  for (int kk = 0; kk < NQ * HD; kk += 32) {
    __syncthreads();
    // A stage: already f16
    for (int it = 0; it < 8; ++it) {
      int idx = tid + it * 256;             // 0..2047 dwords
      int row = idx >> 4, kd = idx & 15;
      Alds[row * 17 + kd] = ep[(size_t)(r0 + row) * 1024 + (kk >> 1) + kd];
    }
    // B stage: wo flat [2048][2048] f32 -> Bt[col][k] f16
    for (int it = 0; it < 4; ++it) {
      int idx = tid + it * 256;
      int i = idx >> 5, h4 = (idx & 31) * 4;
      const float* pw = wo + (size_t)(kk + i) * DN + c0 + h4;
      float4 f = *(const float4*)pw;
      if (kk + 32 < NQ * HD) __builtin_prefetch(pw + 32 * DN, 0, 0);
      h16* bt = (h16*)Btlds;
      bt[(h4 + 0) * 34 + i] = (h16)f.x;
      bt[(h4 + 1) * 34 + i] = (h16)f.y;
      bt[(h4 + 2) * 34 + i] = (h16)f.z;
      bt[(h4 + 3) * 34 + i] = (h16)f.w;
    }
    __syncthreads();

    Frag a;
    const int arow = wave * 16 + l16;
    for (int j = 0; j < 4; ++j) a.u[j]     = Alds[arow * 17 + half * 4 + j];
    for (int j = 0; j < 4; ++j) a.u[4 + j] = Alds[arow * 17 + 8 + half * 4 + j];
    for (int c2 = 0; c2 < 8; ++c2) {
      Frag bf;
      const int col = c2 * 16 + l16;
      for (int j = 0; j < 8; ++j) bf.u[j] = Btlds[col * 17 + (j >> 2) * 8 + half * 4 + (j & 3)];
      acc[c2] = wmma_f16(a, bf, acc[c2]);
    }
  }

  for (int r = 0; r < 8; ++r) {
    const int row = r0 + wave * 16 + half * 8 + r;
    for (int j = 0; j < 8; ++j)
      out[(size_t)row * DN + c0 + j * 16 + l16] = acc[j][r];
  }
}

// =====================================================================
extern "C" void kernel_launch(void* const* d_in, const int* in_sizes, int n_in,
                              void* d_out, int out_size, void* d_ws,
                              size_t ws_size, hipStream_t stream) {
  const float* x      = (const float*)d_in[0];
  const float* wq     = (const float*)d_in[1];
  const float* wkv    = (const float*)d_in[2];
  const float* wo     = (const float*)d_in[3];
  const int* segpos   = (const int*)d_in[4];
  const unsigned char* amask = (const unsigned char*)d_in[5];

  char* ws = (char*)d_ws;
  h16* qh  = (h16*)(ws);                         // 16 MB  [B,N,T,H]
  h16* kh  = (h16*)(ws + (size_t)(16 << 20));    //  8 MB  [B,K,T,H]
  h16* vh  = (h16*)(ws + (size_t)(24 << 20));    //  8 MB  [B,K,T,H]
  h16* enc = (h16*)(ws + (size_t)(32 << 20));    // 16 MB  [B,T,N*H]

  dim3 blk(256);
  dim3 g1(TN / 128, 32, BN);
  qkv_kernel<<<g1, blk, 0, stream>>>(x, wq, wkv, segpos, qh, kh, vh);
  dim3 g2(TN / 128, NQ, BN);
  attn_kernel<<<g2, blk, 0, stream>>>(qh, kh, vh, amask, enc);
  dim3 g3(BN * TN / 128, DN / 128);
  outproj_kernel<<<g3, blk, 0, stream>>>(enc, wo, (float*)d_out);
}
